// HNO_75471165325658
// MI455X (gfx1250) — matrix operand order repaired and checked by reading, not compiled, not run on hardware
//
#include <hip/hip_runtime.h>
#include <hip/hip_bf16.h>

// ---------------- problem constants ----------------
#define Bc       256
#define NPGc     512
#define COMPc    8
#define FEATSc   14
#define HIDc     256
#define NCLSc    21
#define N_NODES  (Bc * NPGc)            // 131072
#define E_CHEB   (Bc * 4 * NPGc * 2)    // 1048576
#define STRIDEg  (NPGc + COMPc)         // 520
#define N2c      (Bc * STRIDEg)         // 133120
#define NLAT     (Bc * COMPc)           // 2048
#define E_PATCH  (Bc * NPGc)            // 131072
#define E_LAT    (Bc * COMPc * COMPc)   // 16384
#define E_GAT1   (E_PATCH + N2c)        // 264192
#define E_REC    (Bc * NPGc)            // 131072
#define E_GAT2   (E_REC + N2c)          // 264192
#define OUTSZ    (N_NODES * NCLSc)

typedef __attribute__((ext_vector_type(16))) __bf16 v16bf;
typedef __attribute__((ext_vector_type(8)))  float  v8f;

#define LDS_PTR(p) ((__attribute__((address_space(3))) char*)(p))

// ---------------- small helpers ----------------
__device__ inline void atomicMaxF(float* addr, float val) {
  int* ia = (int*)addr;
  int old = __float_as_int(*addr);
  while (__int_as_float(old) < val) {
    int assumed = old;
    old = atomicCAS(ia, assumed, __float_as_int(val));
    if (old == assumed) break;
  }
}
__device__ inline unsigned pk_bf16(float lo, float hi) {
  unsigned short a = __builtin_bit_cast(unsigned short, (__bf16)lo);
  unsigned short b = __builtin_bit_cast(unsigned short, (__bf16)hi);
  return (unsigned)a | ((unsigned)b << 16);
}

// =====================================================================
// FAST WMMA GEMM: C[M,N] = A[M,K] @ W[K,N]; requires M%128==0, K%32==0,
// N%64==0.  Block tile 128x64, 8 waves, each wave -> 16 rows x 4 n-tiles
// (4 x v_wmma_f32_16x16x32_bf16 per k-step, A-fragment reused).
// A tiles streamed HBM->LDS with CDNA5 global_load_async_to_lds_b128
// (double-buffered, s_wait_asynccnt); B tiles register-prefetched and
// stored transposed+bf16-packed (one ds_store_b128/thread).
// flags: bit0 = accumulate into C, bit1 = relu.  bias may be null.
// =====================================================================
__global__ __launch_bounds__(256) void gemm_wmma_fast(
    const float* __restrict__ A, const float* __restrict__ W,
    float* __restrict__ C, const float* __restrict__ bias,
    int M, int K, int Nout, int flags)
{
  __shared__ float  lds_af[2][128 * 32];   // A tile, f32 [row][k]
  __shared__ __bf16 lds_b16[2][64 * 32];   // B tile, bf16 [n][k]

  const int m0   = blockIdx.x * 128;
  const int n0   = blockIdx.y * 64;
  const int tid  = threadIdx.x;
  const int wave = tid >> 5;
  const int lane = tid & 31;
  const int wm   = wave * 16;
  // WMMA source layouts (ISA 7.12.2)
  const int arow = lane & 15;
  const int akb  = (lane >> 4) * 8;
  const int bcol = lane & 15;
  const int bkh  = (lane >> 4) * 8;
  // B staging assignment: thread -> (n, 8 consecutive k)
  const int bnn = tid >> 2;
  const int bkb = (tid & 3) * 8;

  v8f acc[4];
#pragma unroll
  for (int nt = 0; nt < 4; ++nt)
    acc[nt] = (v8f){0.f, 0.f, 0.f, 0.f, 0.f, 0.f, 0.f, 0.f};

  // issue async HBM->LDS copies of a 128x32 f32 A tile
  auto asyncA = [&](int k0, int buf) {
#pragma unroll
    for (int c = 0; c < 4; ++c) {
      int fidx = (tid + c * 256) * 4;          // float index within tile
      int r = fidx >> 5, kk = fidx & 31;
      const float* g = &A[(size_t)(m0 + r) * K + k0 + kk];
      asm volatile("global_load_async_to_lds_b128 %0, %1, off"
                   :: "v"(LDS_PTR(&lds_af[buf][fidx])), "v"(g)
                   : "memory");
    }
  };
  auto loadB = [&](int k0, float* brg) {
#pragma unroll
    for (int q = 0; q < 8; ++q)
      brg[q] = W[(size_t)(k0 + bkb + q) * Nout + n0 + bnn];
  };
  auto storeB = [&](int buf, const float* brg) {
    uint4 p;
    p.x = pk_bf16(brg[0], brg[1]);
    p.y = pk_bf16(brg[2], brg[3]);
    p.z = pk_bf16(brg[4], brg[5]);
    p.w = pk_bf16(brg[6], brg[7]);
    *(uint4*)&lds_b16[buf][bnn * 32 + bkb] = p;
  };

  // prologue: stage tile 0
  {
    float brg[8];
    asyncA(0, 0);
    loadB(0, brg);
    storeB(0, brg);
    asm volatile("s_wait_asynccnt 0" ::: "memory");
    __syncthreads();
  }

  for (int k0 = 0; k0 < K; k0 += 32) {
    const int cur = (k0 >> 5) & 1, nxt = cur ^ 1;
    const bool more = (k0 + 32) < K;
    float brg[8];
    if (more) {               // overlap next-tile fetch with this tile's WMMAs
      asyncA(k0 + 32, nxt);
      loadB(k0 + 32, brg);
    }

    // A fragment: f32 from LDS -> bf16 (cvt co-executes with WMMA)
    const float* ap = &lds_af[cur][(wm + arow) * 32];
    v16bf av;
#pragma unroll
    for (int q = 0; q < 4; ++q) {
      av[2 * q]         = (__bf16)ap[akb + 2 * q];
      av[2 * q + 1]     = (__bf16)ap[akb + 2 * q + 1];
      av[8 + 2 * q]     = (__bf16)ap[16 + akb + 2 * q];
      av[8 + 2 * q + 1] = (__bf16)ap[16 + akb + 2 * q + 1];
    }
#pragma unroll
    for (int nt = 0; nt < 4; ++nt) {
      const __bf16* bp = &lds_b16[cur][(nt * 16 + bcol) * 32 + bkh];
      v16bf bv;
#pragma unroll
      for (int j = 0; j < 16; ++j) bv[j] = bp[j < 8 ? j : j + 8];
      acc[nt] = __builtin_amdgcn_wmma_f32_16x16x32_bf16(false, av, false, bv,
                                                        (short)0, acc[nt],
                                                        false, false);
    }

    if (more) storeB(nxt, brg);
    asm volatile("s_wait_asynccnt 0" ::: "memory");
    __syncthreads();
  }

  // C layout: VGPR r, lanes 0-15 -> M=r; lanes 16-31 -> M=8+r; N=lane%16.
  const int crow = (lane >> 4) * 8;
#pragma unroll
  for (int nt = 0; nt < 4; ++nt) {
    const int col = n0 + nt * 16 + (lane & 15);
#pragma unroll
    for (int r = 0; r < 8; ++r) {
      size_t off = (size_t)(m0 + wm + crow + r) * Nout + col;
      float v = acc[nt][r];
      if (flags & 1) v += C[off];
      if (bias) v += bias[col];
      if (flags & 2) v = fmaxf(v, 0.f);
      C[off] = v;
    }
  }
}

// =====================================================================
// RAGGED WMMA GEMM (zero-padded LDS stage) for K=14 / Nout=21 shapes.
// =====================================================================
__global__ __launch_bounds__(256) void gemm_wmma(
    const float* __restrict__ A, const float* __restrict__ W,
    float* __restrict__ C, const float* __restrict__ bias,
    int M, int K, int Nout, int flags)
{
  __shared__ __bf16 lds_a[128 * 32];
  __shared__ __bf16 lds_b[16 * 32];

  const int m0   = blockIdx.x * 128;
  const int n0   = blockIdx.y * 16;
  const int tid  = threadIdx.x;
  const int wave = tid >> 5;
  const int lane = tid & 31;
  const int wm   = wave * 16;

  v8f acc = {0.f, 0.f, 0.f, 0.f, 0.f, 0.f, 0.f, 0.f};
  const int arow = lane & 15;
  const int akb  = (lane >> 4) * 8;
  const int bcol = lane & 15;
  const int bkh  = (lane >> 4) * 8;

  for (int k0 = 0; k0 < K; k0 += 32) {
    if (k0 + 32 < K)
      __builtin_prefetch(&A[(size_t)(m0 + (tid >> 1)) * K + k0 + 32], 0, 0);
#pragma unroll
    for (int i = 0; i < 16; ++i) {
      int idx = tid + i * 256;
      int r = idx >> 5, kk = idx & 31;
      float v = (k0 + kk < K) ? A[(size_t)(m0 + r) * K + k0 + kk] : 0.f;
      lds_a[r * 32 + kk] = (__bf16)v;
    }
    if (tid < 512) {
      int nn = tid >> 5, kk = tid & 31;
      float v = 0.f;
      if ((n0 + nn) < Nout && (k0 + kk) < K) v = W[(size_t)(k0 + kk) * Nout + n0 + nn];
      lds_b[nn * 32 + kk] = (__bf16)v;
    }
    __syncthreads();

    v16bf av, bv;
#pragma unroll
    for (int q = 0; q < 4; ++q) {
      int kk = akb + 2 * q;
      av[2 * q]     = lds_a[(wm + arow) * 32 + kk];
      av[2 * q + 1] = lds_a[(wm + arow) * 32 + kk + 1];
      int kk2 = 16 + akb + 2 * q;
      av[8 + 2 * q]     = lds_a[(wm + arow) * 32 + kk2];
      av[8 + 2 * q + 1] = lds_a[(wm + arow) * 32 + kk2 + 1];
    }
#pragma unroll
    for (int q = 0; q < 8; ++q) {
      int kk = bkh + 2 * q + (q >= 4 ? 8 : 0);
      bv[2 * q]     = lds_b[bcol * 32 + kk];
      bv[2 * q + 1] = lds_b[bcol * 32 + kk + 1];
    }
    acc = __builtin_amdgcn_wmma_f32_16x16x32_bf16(false, av, false, bv,
                                                  (short)0, acc, false, false);
    __syncthreads();
  }

  const int crow = (lane >> 4) * 8;
  const int ccol = n0 + (lane & 15);
  if (ccol < Nout) {
#pragma unroll
    for (int r = 0; r < 8; ++r) {
      size_t off = (size_t)(m0 + wm + crow + r) * Nout + ccol;
      float v = acc[r];
      if (flags & 1) v += C[off];
      if (bias) v += bias[ccol];
      if (flags & 2) v = fmaxf(v, 0.f);
      C[off] = v;
    }
  }
}

// ---------------- graph / elementwise kernels ----------------
__global__ void deg_kernel(const long long* src, float* deg, int E) {
  int e = blockIdx.x * blockDim.x + threadIdx.x;
  if (e < E) atomicAdd(&deg[(int)src[e]], 1.0f);
}
__global__ void dinv_kernel(const float* deg, float* dinv, int n) {
  int i = blockIdx.x * blockDim.x + threadIdx.x;
  if (i < n) { float d = deg[i]; dinv[i] = d > 0.f ? rsqrtf(fmaxf(d, 1e-12f)) : 0.f; }
}
__global__ void edgew_kernel(const long long* s, const long long* d,
                             const float* dinv, float* w, int E) {
  int e = blockIdx.x * blockDim.x + threadIdx.x;
  if (e < E) w[e] = -(dinv[(int)s[e]] * dinv[(int)d[e]]);
}
__global__ void prop_kernel(const long long* src, const long long* dst,
                            const float* w, const float* V, float* O, int F) {
  int e = blockIdx.x, f = threadIdx.x;
  if (f >= F) return;
  int s = (int)src[e], d = (int)dst[e];
  atomicAdd(&O[(size_t)d * F + f], w[e] * V[(size_t)s * F + f]);
}
__global__ void comb_kernel(float* t, const float* x0, int n) {   // t = 2t - x0
  int i = blockIdx.x * blockDim.x + threadIdx.x;
  if (i < n) t[i] = 2.f * t[i] - x0[i];
}
__global__ void relu_kernel(float* x, int n) {
  int i = blockIdx.x * blockDim.x + threadIdx.x;
  if (i < n) x[i] = fmaxf(x[i], 0.f);
}
__global__ __launch_bounds__(256) void bn_stats(const float* X, float* s, float* ss, int M) {
  int col = threadIdx.x;
  int r0 = blockIdx.x * 512, r1 = min(M, r0 + 512);
  float a = 0.f, b = 0.f;
  for (int r = r0; r < r1; ++r) { float v = X[(size_t)r * HIDc + col]; a += v; b += v * v; }
  atomicAdd(&s[col], a); atomicAdd(&ss[col], b);
}
__global__ void bn_norm(float* X, const float* s, const float* ss,
                        const float* g, const float* b, int M, int reluAfter) {
  int idx = blockIdx.x * blockDim.x + threadIdx.x;
  int col = idx & (HIDc - 1);
  float mu = s[col] / (float)M;
  float var = ss[col] / (float)M - mu * mu;
  float y = (X[idx] - mu) * rsqrtf(var + 1e-5f) * g[col] + b[col];
  if (reluAfter) y = fmaxf(y, 0.f);
  X[idx] = y;
}
__global__ void fillv_kernel(float* p, float v, int n) {
  int i = blockIdx.x * blockDim.x + threadIdx.x;
  if (i < n) p[i] = v;
}
__global__ void matvec_kernel(const float* H, const float* a, float* out, int n) {
  int i = blockIdx.x * blockDim.x + threadIdx.x;
  if (i >= n) return;
  float s = 0.f;
  for (int f = 0; f < HIDc; ++f) s += H[(size_t)i * HIDc + f] * a[f];
  out[i] = s;
}
__global__ void build_edges(const long long* s64, const long long* d64,
                            int Epe, int n, int* s32, int* d32) {
  int t = blockIdx.x * blockDim.x + threadIdx.x;
  if (t >= Epe + n) return;
  if (t < Epe) { s32[t] = (int)s64[t]; d32[t] = (int)d64[t]; }
  else { int i = t - Epe; s32[t] = i; d32[t] = i; }
}
__global__ void build_edges_rec(const long long* RA, int* s32, int* d32) {
  int t = blockIdx.x * blockDim.x + threadIdx.x;
  if (t >= E_REC + N2c) return;
  if (t < E_REC) {
    int g = t >> 9, j = t & 511;
    s32[t] = g * STRIDEg + (int)RA[t];
    d32[t] = g * STRIDEg + COMPc + j;
  } else { int i = t - E_REC; s32[t] = i; d32[t] = i; }
}
__global__ void alpha_kernel(const int* src, const int* dst, const float* as_,
                             const float* ad_, float* alpha, int E) {
  int e = blockIdx.x * blockDim.x + threadIdx.x;
  if (e >= E) return;
  float a = as_[src[e]] + ad_[dst[e]];
  alpha[e] = a >= 0.f ? a : 0.2f * a;
}
__global__ void segmax_kernel(const int* dst, const float* alpha, float* m, int E) {
  int e = blockIdx.x * blockDim.x + threadIdx.x;
  if (e < E) atomicMaxF(&m[dst[e]], alpha[e]);
}
__global__ void expden_kernel(const int* dst, const float* alpha, const float* m,
                              float* expe, float* denom, int E) {
  int e = blockIdx.x * blockDim.x + threadIdx.x;
  if (e >= E) return;
  int d = dst[e];
  float ee = __expf(alpha[e] - m[d]);
  expe[e] = ee;
  atomicAdd(&denom[d], ee);
}
__global__ __launch_bounds__(256) void gat_scatter(const int* src, const int* dst,
    const float* expe, const float* denom, const float* H, float* O) {
  int e = blockIdx.x, f = threadIdx.x;
  int s = src[e], d = dst[e];
  float c = expe[e] / denom[d];
  atomicAdd(&O[(size_t)d * HIDc + f], c * H[(size_t)s * HIDc + f]);
}
__global__ void addbias_kernel(float* O, const float* b, int n) {
  int i = blockIdx.x * blockDim.x + threadIdx.x;
  if (i < n) O[i] += b[i & (HIDc - 1)];
}
__global__ void assemble_x2(const float* XC, const float* P, float* X2) {
  int idx = blockIdx.x * blockDim.x + threadIdx.x;   // over N2c*HIDc
  int row = idx >> 8, f = idx & 255;
  int g = row / STRIDEg, p = row % STRIDEg;
  X2[idx] = (p < NPGc) ? XC[((size_t)g * NPGc + p) * HIDc + f]
                       : P[(p - NPGc) * HIDc + f];
}
__global__ void gather_latent(const float* G, float* L) {
  int idx = blockIdx.x * blockDim.x + threadIdx.x;   // over NLAT*HIDc
  int node = idx >> 8, f = idx & 255;
  int g = node >> 3, c = node & 7;
  L[idx] = G[((size_t)g * STRIDEg + NPGc + c) * HIDc + f];
}
__global__ __launch_bounds__(256) void gin_scatter(const long long* s64, const long long* d64,
                                                   const float* L, float* O) {
  int e = blockIdx.x, f = threadIdx.x;
  int s = (int)s64[e], d = (int)d64[e];
  atomicAdd(&O[(size_t)d * HIDc + f], L[(size_t)s * HIDc + f]);
}
__global__ void assemble_big(const float* CN, const float* XC, float* BG) {
  int idx = blockIdx.x * blockDim.x + threadIdx.x;   // over N2c*HIDc
  int row = idx >> 8, f = idx & 255;
  int g = row / STRIDEg, p = row % STRIDEg;
  BG[idx] = (p < COMPc) ? CN[((size_t)g * COMPc + p) * HIDc + f]
                        : XC[((size_t)g * NPGc + (p - COMPc)) * HIDc + f];
}
__global__ void gather_recs(const float* F_, const float* XC, float* R) {
  int idx = blockIdx.x * blockDim.x + threadIdx.x;   // over N_NODES*HIDc
  int row = idx >> 8, f = idx & 255;
  int g = row >> 9, j = row & 511;
  R[idx] = F_[((size_t)g * STRIDEg + COMPc + j) * HIDc + f] + XC[idx];
}

// ---------------- host driver ----------------
extern "C" void kernel_launch(void* const* d_in, const int* in_sizes, int n_in,
                              void* d_out, int out_size, void* d_ws, size_t ws_size,
                              hipStream_t stream) {
  const float*     x        = (const float*)d_in[0];
  const long long* ei       = (const long long*)d_in[1];   // [2][E_CHEB]
  const float*     params   = (const float*)d_in[3];
  const long long* patch_ei = (const long long*)d_in[5];   // [2][E_PATCH]
  const long long* lat_ei   = (const long long*)d_in[6];   // [2][E_LAT]
  const long long* rec_asg  = (const long long*)d_in[7];
  const float* cheb_w[4] = {(const float*)d_in[8],  (const float*)d_in[10],
                            (const float*)d_in[12], (const float*)d_in[14]};
  const float* cheb_b[4] = {(const float*)d_in[9],  (const float*)d_in[11],
                            (const float*)d_in[13], (const float*)d_in[15]};
  const float* bn_g[3] = {(const float*)d_in[16], (const float*)d_in[18], (const float*)d_in[20]};
  const float* bn_b[3] = {(const float*)d_in[17], (const float*)d_in[19], (const float*)d_in[21]};
  const float* gat1_w  = (const float*)d_in[22];
  const float* gat1_as = (const float*)d_in[23];
  const float* gat1_ad = (const float*)d_in[24];
  const float* gat1_b  = (const float*)d_in[25];
  const float* gin_w1  = (const float*)d_in[26];
  const float* gin_b1  = (const float*)d_in[27];
  const float* gin_w2  = (const float*)d_in[28];
  const float* gin_b2  = (const float*)d_in[29];
  const float* gat2_w  = (const float*)d_in[30];
  const float* gat2_as = (const float*)d_in[31];
  const float* gat2_ad = (const float*)d_in[32];
  const float* gat2_b  = (const float*)d_in[33];
  const float* mlp_w0  = (const float*)d_in[34];
  const float* mlp_bng = (const float*)d_in[35];
  const float* mlp_bnb = (const float*)d_in[36];
  const float* mlp_w1  = (const float*)d_in[37];
  const float* mlp_b1  = (const float*)d_in[38];
  float* out = (float*)d_out;

  // ---- workspace carving ----
  float* ws = (float*)d_ws;
  size_t off = 0;
  auto alloc = [&](size_t n) { float* p = ws + off; off += n; return p; };
  float* BIG0 = alloc((size_t)N2c * HIDc);
  float* BIG1 = alloc((size_t)N2c * HIDc);
  float* BIG2 = alloc((size_t)N2c * HIDc);
  float* BIG3 = alloc((size_t)N2c * HIDc);
  float* deg   = alloc(N_NODES);
  float* dinv  = alloc(N_NODES);
  float* wedge = alloc(E_CHEB);
  float* t14a = alloc((size_t)N_NODES * FEATSc);
  float* t14b = alloc((size_t)N_NODES * FEATSc);
  float* t14c = alloc((size_t)N_NODES * FEATSc);
  float* asv   = alloc(N2c);
  float* adv   = alloc(N2c);
  float* mseg  = alloc(N2c);
  float* denom = alloc(N2c);
  float* alpha = alloc(E_GAT1);
  float* expe  = alloc(E_GAT1);
  int*   esrc  = (int*)alloc(E_GAT1);
  int*   edst  = (int*)alloc(E_GAT1);
  float* latent = alloc((size_t)NLAT * HIDc);
  float* hsum   = alloc((size_t)NLAT * HIDc);
  float* ginmid = alloc((size_t)NLAT * HIDc);
  float* compb  = alloc((size_t)NLAT * HIDc);
  float* bnsum = alloc(HIDc);
  float* bnsq  = alloc(HIDc);
  (void)ws_size; (void)in_sizes; (void)n_in; (void)out_size;

  auto run_gemm = [&](const float* A, const float* W, float* C, const float* bias,
                      int M, int K, int Nout, int flags) {
    if ((K & 31) == 0 && (Nout & 63) == 0) {
      dim3 g(M / 128, Nout / 64);
      gemm_wmma_fast<<<g, 256, 0, stream>>>(A, W, C, bias, M, K, Nout, flags);
    } else {
      dim3 g(M / 128, (Nout + 15) / 16);
      gemm_wmma<<<g, 256, 0, stream>>>(A, W, C, bias, M, K, Nout, flags);
    }
  };
  auto run_prop = [&](const float* vin, float* vout, int F) {
    hipMemsetAsync(vout, 0, (size_t)N_NODES * F * sizeof(float), stream);
    int bt = (F + 31) / 32 * 32;
    prop_kernel<<<E_CHEB, bt, 0, stream>>>(ei, ei + E_CHEB, wedge, vin, vout, F);
  };
  auto run_bn = [&](float* X, const float* g, const float* b, int M, int reluAfter) {
    hipMemsetAsync(bnsum, 0, HIDc * sizeof(float), stream);
    hipMemsetAsync(bnsq, 0, HIDc * sizeof(float), stream);
    bn_stats<<<(M + 511) / 512, 256, 0, stream>>>(X, bnsum, bnsq, M);
    bn_norm<<<(M * HIDc) / 256, 256, 0, stream>>>(X, bnsum, bnsq, g, b, M, reluAfter);
  };

  // ---- graph normalization weights (shared by all ChebConvs) ----
  hipMemsetAsync(deg, 0, N_NODES * sizeof(float), stream);
  deg_kernel<<<E_CHEB / 256, 256, 0, stream>>>(ei, deg, E_CHEB);
  dinv_kernel<<<N_NODES / 256, 256, 0, stream>>>(deg, dinv, N_NODES);
  edgew_kernel<<<E_CHEB / 256, 256, 0, stream>>>(ei, ei + E_CHEB, dinv, wedge, E_CHEB);

  // ---- ChebConv layer 1 (FEATS=14 -> 256) ----
  {
    const int W1 = FEATSc * HIDc;
    run_gemm(x, cheb_w[0] + 0 * W1, BIG0, cheb_b[0], N_NODES, FEATSc, HIDc, 0);
    run_prop(x, t14a, FEATSc);
    run_gemm(t14a, cheb_w[0] + 1 * W1, BIG0, nullptr, N_NODES, FEATSc, HIDc, 1);
    run_prop(t14a, t14b, FEATSc);
    comb_kernel<<<(N_NODES * FEATSc) / 256, 256, 0, stream>>>(t14b, x, N_NODES * FEATSc);
    run_gemm(t14b, cheb_w[0] + 2 * W1, BIG0, nullptr, N_NODES, FEATSc, HIDc, 1);
    run_prop(t14b, t14c, FEATSc);
    comb_kernel<<<(N_NODES * FEATSc) / 256, 256, 0, stream>>>(t14c, t14a, N_NODES * FEATSc);
    run_gemm(t14c, cheb_w[0] + 3 * W1, BIG0, nullptr, N_NODES, FEATSc, HIDc, 1);
    relu_kernel<<<(N_NODES * HIDc) / 256, 256, 0, stream>>>(BIG0, N_NODES * HIDc);
    run_bn(BIG0, bn_g[0], bn_b[0], N_NODES, 0);
  }

  // ---- ChebConv layers 2-4 (256 -> 256) with buffer rotation ----
  auto cheb256 = [&](float* Pin, float* Pq, float* Pr, float* Pout,
                     const float* Wt, const float* bt) {
    const int WH = HIDc * HIDc;
    run_gemm(Pin, Wt + 0 * WH, Pout, bt, N_NODES, HIDc, HIDc, 0);
    run_prop(Pin, Pq, HIDc);
    run_gemm(Pq, Wt + 1 * WH, Pout, nullptr, N_NODES, HIDc, HIDc, 1);
    run_prop(Pq, Pr, HIDc);
    comb_kernel<<<(N_NODES * HIDc) / 256, 256, 0, stream>>>(Pr, Pin, N_NODES * HIDc);
    run_gemm(Pr, Wt + 2 * WH, Pout, nullptr, N_NODES, HIDc, HIDc, 1);
    run_prop(Pr, Pin, HIDc);   // Pin reused as Tx3 scratch (zeroed inside)
    comb_kernel<<<(N_NODES * HIDc) / 256, 256, 0, stream>>>(Pin, Pq, N_NODES * HIDc);
    run_gemm(Pin, Wt + 3 * WH, Pout, nullptr, N_NODES, HIDc, HIDc, 1);
  };
  cheb256(BIG0, BIG1, BIG2, BIG3, cheb_w[1], cheb_b[1]);   // out in BIG3
  relu_kernel<<<(N_NODES * HIDc) / 256, 256, 0, stream>>>(BIG3, N_NODES * HIDc);
  run_bn(BIG3, bn_g[1], bn_b[1], N_NODES, 0);
  cheb256(BIG3, BIG0, BIG1, BIG2, cheb_w[2], cheb_b[2]);   // out in BIG2
  relu_kernel<<<(N_NODES * HIDc) / 256, 256, 0, stream>>>(BIG2, N_NODES * HIDc);
  run_bn(BIG2, bn_g[2], bn_b[2], N_NODES, 0);
  cheb256(BIG2, BIG3, BIG0, BIG1, cheb_w[3], cheb_b[3]);   // xcheb = BIG1 (no BN)

  // ---- GAT1 over x2 = [graph nodes | params] ----
  assemble_x2<<<((size_t)N2c * HIDc) / 256, 256, 0, stream>>>(BIG1, params, BIG0);
  run_gemm(BIG0, gat1_w, BIG2, nullptr, N2c, HIDc, HIDc, 0);   // h1 = BIG2
  matvec_kernel<<<(N2c + 255) / 256, 256, 0, stream>>>(BIG2, gat1_as, asv, N2c);
  matvec_kernel<<<(N2c + 255) / 256, 256, 0, stream>>>(BIG2, gat1_ad, adv, N2c);
  build_edges<<<(E_GAT1 + 255) / 256, 256, 0, stream>>>(patch_ei, patch_ei + E_PATCH,
                                                        E_PATCH, N2c, esrc, edst);
  alpha_kernel<<<(E_GAT1 + 255) / 256, 256, 0, stream>>>(esrc, edst, asv, adv, alpha, E_GAT1);
  fillv_kernel<<<(N2c + 255) / 256, 256, 0, stream>>>(mseg, -1e30f, N2c);
  hipMemsetAsync(denom, 0, N2c * sizeof(float), stream);
  segmax_kernel<<<(E_GAT1 + 255) / 256, 256, 0, stream>>>(edst, alpha, mseg, E_GAT1);
  expden_kernel<<<(E_GAT1 + 255) / 256, 256, 0, stream>>>(edst, alpha, mseg, expe, denom, E_GAT1);
  hipMemsetAsync(BIG3, 0, (size_t)N2c * HIDc * sizeof(float), stream);
  gat_scatter<<<E_GAT1, 256, 0, stream>>>(esrc, edst, expe, denom, BIG2, BIG3);
  addbias_kernel<<<((size_t)N2c * HIDc) / 256, 256, 0, stream>>>(BIG3, gat1_b, N2c * HIDc);
  gather_latent<<<((size_t)NLAT * HIDc) / 256, 256, 0, stream>>>(BIG3, latent);

  // ---- GIN over per-graph complete latent graphs ----
  hipMemcpyAsync(hsum, latent, (size_t)NLAT * HIDc * sizeof(float),
                 hipMemcpyDeviceToDevice, stream);
  gin_scatter<<<E_LAT, 256, 0, stream>>>(lat_ei, lat_ei + E_LAT, latent, hsum);
  run_gemm(hsum, gin_w1, ginmid, gin_b1, NLAT, HIDc, HIDc, 2);   // relu(h@w1+b1)
  run_gemm(ginmid, gin_w2, compb, gin_b2, NLAT, HIDc, HIDc, 0);

  // ---- GAT2: reconstruct latent -> nodes ----
  assemble_big<<<((size_t)N2c * HIDc) / 256, 256, 0, stream>>>(compb, BIG1, BIG0);
  run_gemm(BIG0, gat2_w, BIG2, nullptr, N2c, HIDc, HIDc, 0);    // h2 = BIG2
  matvec_kernel<<<(N2c + 255) / 256, 256, 0, stream>>>(BIG2, gat2_as, asv, N2c);
  matvec_kernel<<<(N2c + 255) / 256, 256, 0, stream>>>(BIG2, gat2_ad, adv, N2c);
  build_edges_rec<<<(E_GAT2 + 255) / 256, 256, 0, stream>>>(rec_asg, esrc, edst);
  alpha_kernel<<<(E_GAT2 + 255) / 256, 256, 0, stream>>>(esrc, edst, asv, adv, alpha, E_GAT2);
  fillv_kernel<<<(N2c + 255) / 256, 256, 0, stream>>>(mseg, -1e30f, N2c);
  hipMemsetAsync(denom, 0, N2c * sizeof(float), stream);
  segmax_kernel<<<(E_GAT2 + 255) / 256, 256, 0, stream>>>(edst, alpha, mseg, E_GAT2);
  expden_kernel<<<(E_GAT2 + 255) / 256, 256, 0, stream>>>(edst, alpha, mseg, expe, denom, E_GAT2);
  hipMemsetAsync(BIG3, 0, (size_t)N2c * HIDc * sizeof(float), stream);
  gat_scatter<<<E_GAT2, 256, 0, stream>>>(esrc, edst, expe, denom, BIG2, BIG3);
  addbias_kernel<<<((size_t)N2c * HIDc) / 256, 256, 0, stream>>>(BIG3, gat2_b, N2c * HIDc);

  // recs = final[node_pos] + xcheb  -> BIG2 (h2 no longer needed)
  gather_recs<<<((size_t)N_NODES * HIDc) / 256, 256, 0, stream>>>(BIG3, BIG1, BIG2);

  // ---- MLP head: Linear(no bias) -> BN -> ReLU -> Linear(bias) ----
  run_gemm(BIG2, mlp_w0, BIG0, nullptr, N_NODES, HIDc, HIDc, 0);
  run_bn(BIG0, mlp_bng, mlp_bnb, N_NODES, 1);
  run_gemm(BIG0, mlp_w1, out, mlp_b1, N_NODES, HIDc, NCLSc, 0);

  // tuple (out, out, out)
  hipMemcpyAsync(out + OUTSZ,     out, (size_t)OUTSZ * sizeof(float),
                 hipMemcpyDeviceToDevice, stream);
  hipMemcpyAsync(out + 2 * OUTSZ, out, (size_t)OUTSZ * sizeof(float),
                 hipMemcpyDeviceToDevice, stream);
}
